// TMAHeadForce_75565654606021
// MI455X (gfx1250) — matrix operand order repaired and checked by reading, not compiled, not run on hardware
//
#include <hip/hip_runtime.h>
#include <stdint.h>

// ---------------------------------------------------------------------------
// MI455X (gfx1250) implementation.
// All GEMM-shaped work (1x1 convs, implicit 3x3 convs, QK^T, attn*V) runs on
// v_wmma_f32_16x16x32_bf16 with f32 accumulation. Activations/weights are
// converted once to bf16. Weights (and the weighted-V matrix) are pre-swizzled
// in global memory into the exact WMMA A-fragment layout so each 128x32 A tile
// is one contiguous 8KB blob that the Tensor Data Mover copies straight into
// LDS (tensor_load_to_lds + s_wait_tensorcnt), double-buffered against the
// WMMA compute. B tiles are register-staged and stored to LDS pre-swizzled in
// B-fragment layout (packed b32 stores), so every lane fetches its fragments
// with ds_load_b128. Workspace requirement: ~406 MiB.
// ---------------------------------------------------------------------------

typedef __attribute__((ext_vector_type(16))) __bf16 v16bf;
typedef __attribute__((ext_vector_type(8)))  float  v8f;

#define TT   2
#define BB   2
#define CIN  2048
#define HWSZ 4096           // H*W (H=W=64)
#define CK   256
#define CV   512
#define COUT 512
#define MATT 8192           // T*H*W

union FragU { uint4 q[2]; v16bf v; };

__device__ __forceinline__ unsigned short f2bf(float f) {
    unsigned u = __float_as_uint(f);
    u = u + 0x7FFFu + ((u >> 16) & 1u);   // round-to-nearest-even
    return (unsigned short)(u >> 16);
}
__device__ __forceinline__ float bf2f(unsigned short h) {
    return __uint_as_float(((unsigned)h) << 16);
}

// ---- Tensor Data Mover: contiguous 8KB global -> LDS copy -----------------
// This toolchain exposes the 6-arg builtin:
//   (uint32x4 g0, int32x8 g1, int32x4 g2, int32x4 g3, int32x8 extra, i32 cpol)
#if defined(__HIP_DEVICE_COMPILE__) && __has_builtin(__builtin_amdgcn_tensor_load_to_lds)
#define HAVE_TDM 1
typedef unsigned int u32x4 __attribute__((ext_vector_type(4)));
typedef int          i32x4 __attribute__((ext_vector_type(4)));
typedef int          i32x8 __attribute__((ext_vector_type(8)));

__device__ __forceinline__ void tdm_load_8kb(const unsigned short* gsrc, unsigned lds_off) {
    unsigned long long ga = (unsigned long long)(uintptr_t)gsrc;
    u32x4 g0;
    g0.x = 1u;                                         // count=1 (valid user D#)
    g0.y = lds_off;                                    // lds_addr
    g0.z = (unsigned)(ga & 0xFFFFFFFFull);             // global_addr[31:0]
    g0.w = (unsigned)((ga >> 32) & 0x1FFFFFFull) | (2u << 30); // [56:32] | type=2
    i32x8 g1;
    g1[0] = (3 << 16);               // workgroup_mask=0, data_size=3 (8B)
    g1[1] = (int)(1024u << 16);      // tensor_dim0[15:0] = 1024
    g1[2] = (int)(1u << 16);         // tensor_dim0[31:16]=0 | tensor_dim1=1
    g1[3] = (int)(1024u << 16);      // tensor_dim1[31:16]=0 | tile_dim0=1024
    g1[4] = 0;                       // tile_dim1=0 (unused), tile_dim2=0
    g1[5] = 1024;                    // tensor_dim0_stride[31:0]
    g1[6] = 0;
    g1[7] = 0;
    i32x4 z4 = {0, 0, 0, 0};
    i32x8 z8 = {0, 0, 0, 0, 0, 0, 0, 0};
    __builtin_amdgcn_tensor_load_to_lds(g0, g1, z4, z4, z8, 0);
}
#else
#define HAVE_TDM 0
#endif

// ---------------- f32 -> bf16 (vectorized, grid-stride) --------------------
__global__ void cvt_bf16_kernel(const float* __restrict__ src,
                                unsigned short* __restrict__ dst, long n4) {
    long i = (long)blockIdx.x * blockDim.x + threadIdx.x;
    long stride = (long)gridDim.x * blockDim.x;
    for (; i < n4; i += stride) {
        float4 f = ((const float4*)src)[i];
        ushort4 o;
        o.x = f2bf(f.x); o.y = f2bf(f.y); o.z = f2bf(f.z); o.w = f2bf(f.w);
        ((ushort4*)dst)[i] = o;
    }
}

// ---- f32 [M,K] row-major -> bf16 pre-swizzled WMMA A-fragment layout ------
// Output index o = ((ks*(M/16) + ms)*32 + lane)*16 + e, where
// k = ks*32 + (e&7) + 16*(e>>3) + 8*(lane>>4), m = ms*16 + (lane&15).
__global__ void cvt_swz_kernel(const float* __restrict__ src,
                               unsigned short* __restrict__ dst,
                               int M, int K) {
    long total = (long)M * K;
    int msubs = M >> 4;
    long i = (long)blockIdx.x * blockDim.x + threadIdx.x;
    long stride = (long)gridDim.x * blockDim.x;
    for (; i < total; i += stride) {
        int  e    = (int)(i & 15);
        int  lane = (int)((i >> 4) & 31);
        long blk  = i >> 9;
        int  ms   = (int)(blk % msubs);
        int  ks   = (int)(blk / msubs);
        int  m    = ms * 16 + (lane & 15);
        int  k    = ks * 32 + (e & 7) + ((e >> 3) << 4) + ((lane >> 4) << 3);
        dst[i] = f2bf(src[(long)m * K + k]);
    }
}

// ------- repack (T,B,C,HW) bf16 -> (C, T*HW) for one batch b ---------------
// swz=0: linear [C][T*HW] (used as a B operand).
// swz=1: WMMA A-fragment pre-swizzled layout (M=C rows, K=T*HW), msubs=C/16,
//        with the per-frame force weights folded in (w_min t=0, w_max t=1).
__global__ void pack_tb_kernel(const unsigned short* __restrict__ src,
                               unsigned short* __restrict__ dst,
                               int C, int b, int Bn,
                               const float* __restrict__ force, int apply_w,
                               int swz) {
    float w0 = 1.f, w1 = 1.f;
    if (apply_w) {
        float f0 = force[b * 18 + 2];       // f[b,0,2]
        float f1 = force[b * 18 + 8];       // f[b,1,2]
        float f2 = force[b * 18 + 14];      // f[b,2,2]
        float dmm = f0 - f1;
        w0 = (f0 - f2) / dmm;
        w1 = (f2 - f1) / dmm;
    }
    long total = (long)C * 2 * HWSZ;
    long i = (long)blockIdx.x * blockDim.x + threadIdx.x;
    long stride = (long)gridDim.x * blockDim.x;
    int msubs = C >> 4;
    for (; i < total; i += stride) {
        long c; int t, p;
        if (swz) {
            int  e    = (int)(i & 15);
            int  lane = (int)((i >> 4) & 31);
            long blk  = i >> 9;
            int  ms   = (int)(blk % msubs);
            int  ks   = (int)(blk / msubs);
            c = ms * 16 + (lane & 15);
            int m = ks * 32 + (e & 7) + ((e >> 3) << 4) + ((lane >> 4) << 3);
            t = m >> 12; p = m & 4095;
        } else {
            c = i >> 13;
            int r = (int)(i & 8191);
            t = r >> 12; p = r & 4095;
        }
        unsigned short v = src[(((long)(t * Bn + b)) * C + c) * HWSZ + p];
        if (apply_w) v = f2bf(bf2f(v) * (t ? w1 : w0));
        dst[i] = v;
    }
}

// ---------------- row softmax: 8192 f32 -> 8192 bf16 -----------------------
__global__ void softmax_kernel(const float* __restrict__ logits,
                               unsigned short* __restrict__ attn, int cols) {
    __shared__ float red[256];
    int row = blockIdx.x;
    const float* L = logits + (long)row * cols;
    const int per = 32;                      // 8192 / 256
    float v[32];
    float m = -1e30f;
    for (int i = 0; i < per; i++) {
        v[i] = L[threadIdx.x + (i << 8)];
        m = fmaxf(m, v[i]);
    }
    red[threadIdx.x] = m; __syncthreads();
    for (int s = 128; s > 0; s >>= 1) {
        if ((int)threadIdx.x < s) red[threadIdx.x] = fmaxf(red[threadIdx.x], red[threadIdx.x + s]);
        __syncthreads();
    }
    m = red[0]; __syncthreads();
    float sum = 0.f;
    for (int i = 0; i < per; i++) { v[i] = __expf(v[i] - m); sum += v[i]; }
    red[threadIdx.x] = sum; __syncthreads();
    for (int s = 128; s > 0; s >>= 1) {
        if ((int)threadIdx.x < s) red[threadIdx.x] += red[threadIdx.x + s];
        __syncthreads();
    }
    float inv = 1.0f / red[0];
    unsigned short* O = attn + (long)row * cols;
    for (int i = 0; i < per; i++) O[threadIdx.x + (i << 8)] = f2bf(v[i] * inv);
}

// ---------------------------------------------------------------------------
// Generic WMMA GEMM: C[M,N] = A[M,K] * B[K,N], block tile 128x128, BK=32,
// double-buffered LDS. 8 waves: wave (wm,wn) owns 64x32 = 4x2 accumulators.
// A: a_swz=1 -> A is pre-swizzled fragment layout (msubs = Mtot/16); the
//    128x32 tile is 8KB contiguous, loaded by wave 0 via tensor_load_to_lds.
//    a_swz=0 -> generic strides (m,k) -> A[m*sa_m + k*sa_k], register-staged.
// B: (k,n) -> B[k*sb_k + n*sb_n], or implicit im2col of a 3x3 conv over a
//    (conv3_cin,64,64) bf16 NCHW image (zero padded) when conv3_cin > 0.
// Epilogue: optional per-row scale/bias + ReLU; bf16 or f32 output with row
// stride ldo. blockIdx.z batches images via strideB_img / strideO_img.
// M, N multiples of 128; K multiple of 32 (true for every layer here).
// ---------------------------------------------------------------------------
__global__ __launch_bounds__(256)
void wmma_gemm_kernel(const unsigned short* __restrict__ A, long sa_m, long sa_k,
                      int a_swz, int msubs,
                      const unsigned short* __restrict__ Bsrc, long sb_k, long sb_n,
                      int conv3_cin, int Kdim,
                      long strideB_img, long strideO_img,
                      const float* __restrict__ scale, const float* __restrict__ bias,
                      int relu,
                      float* __restrict__ outF, unsigned short* __restrict__ outB,
                      long ldo) {
    __shared__ __align__(16) unsigned short As[2][4096];   // 2 x 8KB
    __shared__ __align__(16) unsigned short Bs[2][4096];   // 2 x 8KB

    const int tid  = threadIdx.x;
    const int lane = tid & 31;
    const int wid  = tid >> 5;
    const int wm   = wid >> 2;          // 0..1 (M direction)
    const int wn   = wid & 3;           // 0..3 (N direction)
    const long m0  = (long)blockIdx.y * 128;
    const long n0  = (long)blockIdx.x * 128;
    const unsigned short* Bimg = Bsrc + (long)blockIdx.z * strideB_img;
    const long obase = (long)blockIdx.z * strideO_img;

    v8f acc[4][2];
    #pragma unroll
    for (int i = 0; i < 4; i++)
        #pragma unroll
        for (int j = 0; j < 2; j++)
            acc[i][j] = (v8f){0.f, 0.f, 0.f, 0.f, 0.f, 0.f, 0.f, 0.f};

    // generic-A staging: thread -> (row 0..127, k-half 0/16)
    const int a_m  = tid >> 1;
    const int a_kh = (tid & 1) << 4;
    // B staging: thread -> (2 adjacent k rows, 8 n columns)
    const int b_kp = (tid >> 4) << 1;       // even k row 0..30
    const int b_nc = (tid & 15) << 3;       // n chunk base 0..120

    unsigned short avr[16];
    unsigned short br0[8], br1[8];

#if HAVE_TDM
    const unsigned as_lds0 = (unsigned)(unsigned long long)(uintptr_t)&As[0][0];
#endif

    auto a_issue = [&](int kt, int buf) {
        if (a_swz) {
#if HAVE_TDM
            if (wid == 0)
                tdm_load_8kb(A + (((long)(kt >> 5) * msubs + (m0 >> 4)) << 9),
                             as_lds0 + (unsigned)buf * 8192u);
#else
            const unsigned short* ap =
                A + (((long)(kt >> 5) * msubs + (m0 >> 4)) << 9) + tid * 16;
            *(uint4*)&avr[0] = ((const uint4*)ap)[0];
            *(uint4*)&avr[8] = ((const uint4*)ap)[1];
#endif
        } else {
            const unsigned short* ap = A + (m0 + a_m) * sa_m + (long)(kt + a_kh) * sa_k;
            if (sa_k == 1) {
                *(uint4*)&avr[0] = ((const uint4*)ap)[0];
                *(uint4*)&avr[8] = ((const uint4*)ap)[1];
            } else {
                #pragma unroll
                for (int i = 0; i < 16; i++) avr[i] = ap[(long)i * sa_k];
            }
        }
    };
    auto a_commit = [&](int buf) {          // register -> LDS (non-TDM paths)
        if (a_swz) {
#if !HAVE_TDM
            *(uint4*)&As[buf][tid * 16]     = *(uint4*)&avr[0];
            *(uint4*)&As[buf][tid * 16 + 8] = *(uint4*)&avr[8];
#endif
        } else {
            #pragma unroll
            for (int p = 0; p < 8; p++) {
                int kl   = a_kh + 2 * p;
                int half = (kl >> 3) & 1;
                int e    = (kl & 7) + ((kl >> 4) << 3);    // even
                int ln   = (a_m & 15) + (half << 4);
                int ms   = a_m >> 4;
                *(unsigned*)&As[buf][ms * 512 + ln * 16 + e] =
                    (unsigned)avr[2 * p] | ((unsigned)avr[2 * p + 1] << 16);
            }
        }
    };
    auto b_issue = [&](int kt) {
        if (conv3_cin) {
            #pragma unroll
            for (int r = 0; r < 2; r++) {
                int k  = kt + b_kp + r;
                int ci = k / 9;
                int j  = k - ci * 9;
                int dy = j / 3 - 1;
                int dx = (j - (j / 3) * 3) - 1;
                const unsigned short* inp = Bimg + (long)ci * HWSZ;
                unsigned short* dv = r ? br1 : br0;
                #pragma unroll
                for (int i = 0; i < 8; i++) {
                    int q = (int)n0 + b_nc + i;
                    int y = (q >> 6) + dy;                 // H = W = 64
                    int x = (q & 63) + dx;
                    dv[i] = ((unsigned)y < 64u && (unsigned)x < 64u)
                                ? inp[(y << 6) + x] : (unsigned short)0;
                }
            }
        } else {
            const unsigned short* bp =
                Bimg + (long)(kt + b_kp) * sb_k + (n0 + b_nc) * sb_n;
            if (sb_n == 1) {
                *(uint4*)&br0[0] = *(const uint4*)bp;
                *(uint4*)&br1[0] = *(const uint4*)(bp + sb_k);
            } else {
                #pragma unroll
                for (int i = 0; i < 8; i++) {
                    br0[i] = bp[(long)i * sb_n];
                    br1[i] = bp[sb_k + (long)i * sb_n];
                }
            }
        }
    };
    auto b_commit = [&](int buf) {
        // B-fragment layout: lane = (n&15) + 16*((k>>3)&1), elem = (k&7) + 8*(k>>4).
        int ns = b_nc >> 4;
        int lb = ((b_kp >> 3) & 1) << 4;
        int e  = (b_kp & 7) + ((b_kp >> 4) << 3);   // even -> aligned b32 store
        #pragma unroll
        for (int i = 0; i < 8; i++) {
            int lnn = ((b_nc & 15) + i) + lb;
            *(unsigned*)&Bs[buf][ns * 512 + lnn * 16 + e] =
                (unsigned)br0[i] | ((unsigned)br1[i] << 16);
        }
    };
    auto compute = [&](int buf) {
        FragU af[4], bfm[2];
        #pragma unroll
        for (int im = 0; im < 4; im++) {
            int ms = wm * 4 + im;
            af[im].q[0] = *(const uint4*)&As[buf][ms * 512 + lane * 16];
            af[im].q[1] = *(const uint4*)&As[buf][ms * 512 + lane * 16 + 8];
        }
        #pragma unroll
        for (int in_ = 0; in_ < 2; in_++) {
            int ns = wn * 2 + in_;
            bfm[in_].q[0] = *(const uint4*)&Bs[buf][ns * 512 + lane * 16];
            bfm[in_].q[1] = *(const uint4*)&Bs[buf][ns * 512 + lane * 16 + 8];
        }
        #pragma unroll
        for (int im = 0; im < 4; im++)
            #pragma unroll
            for (int in_ = 0; in_ < 2; in_++)
                acc[im][in_] = __builtin_amdgcn_wmma_f32_16x16x32_bf16(
                    false, af[im].v, false, bfm[in_].v,
                    (short)0, acc[im][in_], false, false);
    };

    // ---- prologue: stage tile 0 into buffer 0 ----
    a_issue(0, 0);
    a_commit(0);
    b_issue(0);
    b_commit(0);
#if HAVE_TDM
    if (a_swz && wid == 0) __builtin_amdgcn_s_wait_tensorcnt(0);
#endif
    __syncthreads();

    int cur = 0;
    for (int kt = 0; kt < Kdim; kt += 32) {
        int nxt = cur ^ 1;
        bool more = (kt + 32) < Kdim;
        if (more) {            // start global traffic for the next tile
            a_issue(kt + 32, nxt);
            b_issue(kt + 32);
        }
        compute(cur);          // 8 WMMAs out of buffer `cur`
        if (more) {
            a_commit(nxt);
            b_commit(nxt);
#if HAVE_TDM
            if (a_swz && wid == 0) __builtin_amdgcn_s_wait_tensorcnt(0);
#endif
        }
        __syncthreads();       // one barrier per K-step
        cur = nxt;
    }

    // ---- epilogue: C layout = VGPR v -> row v + 8*(lane>>4), col lane&15 ---
    const int hi = lane >> 4, lo = lane & 15;
    #pragma unroll
    for (int im = 0; im < 4; im++) {
        long mrow_base = m0 + wm * 64 + im * 16;
        #pragma unroll
        for (int in_ = 0; in_ < 2; in_++) {
            long ncol = n0 + wn * 32 + in_ * 16 + lo;
            #pragma unroll
            for (int v = 0; v < 8; v++) {
                long mrow = mrow_base + v + 8 * hi;
                float val = acc[im][in_][v];
                if (scale) val = val * scale[mrow] + bias[mrow];
                if (relu)  val = fmaxf(val, 0.f);
                long off = obase + mrow * ldo + ncol;
                if (outF) outF[off] = val;
                else      outB[off] = f2bf(val);
            }
        }
    }
}

// ---------------------------------------------------------------------------
extern "C" void kernel_launch(void* const* d_in, const int* in_sizes, int n_in,
                              void* d_out, int out_size, void* d_ws, size_t ws_size,
                              hipStream_t stream) {
    const float* inputs = (const float*)d_in[0];
    const float* seq    = (const float*)d_in[1];
    const float* force  = (const float*)d_in[2];
    const float* Wf[9]; const float* Sf[9]; const float* Bf[9];
    for (int i = 0; i < 9; i++) {
        Wf[i] = (const float*)d_in[3 + 3 * i];
        Sf[i] = (const float*)d_in[4 + 3 * i];
        Bf[i] = (const float*)d_in[5 + 3 * i];
    }
    float* out = (float*)d_out;

    // bump allocator on workspace (256B aligned)
    char* wsb = (char*)d_ws;
    size_t cur = 0;
    auto alloc = [&](size_t bytes) -> void* {
        void* p = wsb + cur;
        cur += (bytes + 255) & ~(size_t)255;
        return p;
    };

    typedef unsigned short u16;
    const long nSeq = (long)TT * BB * CIN * HWSZ;
    const long nIn  = (long)BB * CIN * HWSZ;
    const int  wM[9] = { CK, CK, CV, CV, CK, CK, CV, CV, COUT };
    const int  wK[9] = { CIN, CK*9, CIN, CV*9, CIN, CK*9, CIN, CV*9, 2*CV*9 };

    u16* seq_bf = (u16*)alloc(nSeq * 2);
    u16* in_bf  = (u16*)alloc(nIn * 2);
    u16* wbf[9];
    for (int i = 0; i < 9; i++) wbf[i] = (u16*)alloc((long)wM[i] * wK[i] * 2);
    u16* mk1o   = (u16*)alloc((long)TT*BB*CK*HWSZ * 2);
    u16* mko    = (u16*)alloc((long)TT*BB*CK*HWSZ * 2);
    u16* mv1o   = (u16*)alloc((long)TT*BB*CV*HWSZ * 2);
    u16* mvo    = (u16*)alloc((long)TT*BB*CV*HWSZ * 2);
    u16* qk1o   = (u16*)alloc((long)BB*CK*HWSZ * 2);
    u16* qko    = (u16*)alloc((long)BB*CK*HWSZ * 2);
    u16* qv1o   = (u16*)alloc((long)BB*CV*HWSZ * 2);
    u16* concat = (u16*)alloc((long)BB*2*CV*HWSZ * 2);
    u16* mkpack = (u16*)alloc((long)CK*MATT * 2);      // reused per batch
    u16* mvpack = (u16*)alloc((long)CV*MATT * 2);      // reused per batch (swizzled)
    float* logits = (float*)alloc((long)HWSZ*MATT * 4);// reused per batch
    u16* attn   = (u16*)alloc((long)HWSZ*MATT * 2);    // reused per batch

    auto cvt = [&](const float* s, u16* d, long n) {
        cvt_bf16_kernel<<<4096, 256, 0, stream>>>(s, d, n >> 2);
    };
    auto gemm = [&](const u16* A, long sam, long sak, int aswz, int msubs,
                    const u16* Bp, long sbk, long sbn, int conv3,
                    int M, int N, int K, long sBimg, long sOimg, int nz,
                    const float* sc, const float* bi, int relu,
                    float* oF, u16* oB, long ldo) {
        dim3 g(N / 128, M / 128, nz);
        wmma_gemm_kernel<<<g, 256, 0, stream>>>(A, sam, sak, aswz, msubs,
                                                Bp, sbk, sbn, conv3, K,
                                                sBimg, sOimg, sc, bi, relu, oF, oB, ldo);
    };

    // ---- conversions: activations linear bf16, weights pre-swizzled ----
    cvt(seq, seq_bf, nSeq);
    cvt(inputs, in_bf, nIn);
    for (int i = 0; i < 9; i++)
        cvt_swz_kernel<<<2048, 256, 0, stream>>>(Wf[i], wbf[i], wM[i], wK[i]);

    const long cHW = HWSZ;
    // ---- conv stacks (implicit GEMM, TDM-fed A, z batches images) ----
    gemm(wbf[0], 0,0, 1, CK/16, seq_bf, cHW, 1, 0, CK, HWSZ, CIN,
         (long)CIN*cHW, (long)CK*cHW, TT*BB, Sf[0], Bf[0], 1, nullptr, mk1o, cHW);
    gemm(wbf[1], 0,0, 1, CK/16, mk1o, 0, 0, CK, CK, HWSZ, CK*9,
         (long)CK*cHW, (long)CK*cHW, TT*BB, Sf[1], Bf[1], 1, nullptr, mko, cHW);
    gemm(wbf[2], 0,0, 1, CV/16, seq_bf, cHW, 1, 0, CV, HWSZ, CIN,
         (long)CIN*cHW, (long)CV*cHW, TT*BB, Sf[2], Bf[2], 1, nullptr, mv1o, cHW);
    gemm(wbf[3], 0,0, 1, CV/16, mv1o, 0, 0, CV, CV, HWSZ, CV*9,
         (long)CV*cHW, (long)CV*cHW, TT*BB, Sf[3], Bf[3], 1, nullptr, mvo, cHW);
    gemm(wbf[4], 0,0, 1, CK/16, in_bf, cHW, 1, 0, CK, HWSZ, CIN,
         (long)CIN*cHW, (long)CK*cHW, BB, Sf[4], Bf[4], 1, nullptr, qk1o, cHW);
    gemm(wbf[5], 0,0, 1, CK/16, qk1o, 0, 0, CK, CK, HWSZ, CK*9,
         (long)CK*cHW, (long)CK*cHW, BB, Sf[5], Bf[5], 1, nullptr, qko, cHW);
    gemm(wbf[6], 0,0, 1, CV/16, in_bf, cHW, 1, 0, CV, HWSZ, CIN,
         (long)CIN*cHW, (long)CV*cHW, BB, Sf[6], Bf[6], 1, nullptr, qv1o, cHW);
    gemm(wbf[7], 0,0, 1, CV/16, qv1o, 0, 0, CV, CV, HWSZ, CV*9,
         (long)CV*cHW, (long)2*CV*cHW, BB, Sf[7], Bf[7], 1, nullptr, concat, cHW);

    // ---- attention, per batch ----
    for (int b = 0; b < BB; b++) {
        pack_tb_kernel<<<2048, 256, 0, stream>>>(mko, mkpack, CK, b, BB, nullptr, 0, 0);
        pack_tb_kernel<<<2048, 256, 0, stream>>>(mvo, mvpack, CV, b, BB, force, 1, 1);
        // logits[q,m] = sum_c qk[c,q] * mkf[c,m]  (A read transposed via strides)
        gemm(qko + (long)b*CK*cHW, 1, cHW, 0, 0, mkpack, MATT, 1, 0,
             HWSZ, MATT, CK, 0, 0, 1, nullptr, nullptr, 0, logits, nullptr, MATT);
        softmax_kernel<<<HWSZ, 256, 0, stream>>>(logits, attn, MATT);
        // mem[c,q] = sum_m mvw[c,m] * attn[q,m]  -> concat channels [CV,2CV)
        gemm(mvpack, 0, 0, 1, CV/16, attn, 1, MATT, 0,
             CV, HWSZ, MATT, 0, 0, 1, nullptr, nullptr, 0,
             nullptr, concat + ((long)b*2*CV + CV)*cHW, cHW);
    }

    // ---- final 3x3 conv over concat (1024 -> 512), f32 output ----
    gemm(wbf[8], 0,0, 1, COUT/16, concat, 0, 0, 2*CV, COUT, HWSZ, 2*CV*9,
         (long)2*CV*cHW, (long)COUT*cHW, BB, Sf[8], Bf[8], 1, out, nullptr, cHW);
}
